// PWAM_41781441856050
// MI455X (gfx1250) — matrix operand order repaired and checked by reading, not compiled, not run on hardware
//
#include <hip/hip_runtime.h>
#include <hip/hip_bf16.h>
#include <math.h>

// ---------------- types ----------------
typedef unsigned short u16;
typedef __attribute__((ext_vector_type(16))) __bf16 v16bf;
typedef __attribute__((ext_vector_type(8)))  float  v8f;

union V16 { v16bf v; unsigned u[8]; uint4 q[2]; };
union C8  { v8f v; float f[8]; };

// ---------------- constants ----------------
// B=16, HW=3600, D=KC=VC=512, LC=768, NL=20, H=8, hd=64
#define NB   16
#define NHW  3600
#define ND   512
#define NR   (NB*NHW)          // 57600 rows, = 450 * 128
#define NLC  768
#define NNL  20
#define LDST 40                // LDS row stride in bf16: 80B = 20 banks, 16B-aligned rows

// ---------------- helpers ----------------
__device__ __forceinline__ u16 f2bf(float f) {
    unsigned u = __float_as_uint(f);
    u += 0x7FFFu + ((u >> 16) & 1u);     // RNE
    return (u16)(u >> 16);
}
__device__ __forceinline__ float bflo(unsigned u) { return __uint_as_float(u << 16); }
__device__ __forceinline__ float bfhi(unsigned u) { return __uint_as_float(u & 0xFFFF0000u); }
__device__ __forceinline__ float gelu_exact(float x) {
    return 0.5f * x * (1.0f + erff(x * 0.70710678118654752f));
}

// ---------------- f32 -> bf16 convert (4 elems / thread) ----------------
__global__ __launch_bounds__(256) void cvt_kernel(const float* __restrict__ src,
                                                  u16* __restrict__ dst, int n) {
    int i = (blockIdx.x * 256 + threadIdx.x) * 4;
    if (i >= n) return;
    float4 f = *(const float4*)(src + i);
    unsigned lo = (unsigned)f2bf(f.x) | ((unsigned)f2bf(f.y) << 16);
    unsigned hi = (unsigned)f2bf(f.z) | ((unsigned)f2bf(f.w) << 16);
    uint2 o; o.x = lo; o.y = hi;
    *(uint2*)(dst + i) = o;
}

// ---------------- k / v projections (tiny GEMM, VALU) ----------------
__global__ __launch_bounds__(256) void kv_kernel(const float* __restrict__ l,
                                                 const float* __restrict__ l_mask,
                                                 const float* __restrict__ Wk,
                                                 const float* __restrict__ bk,
                                                 const float* __restrict__ Wv,
                                                 const float* __restrict__ bv,
                                                 float* __restrict__ kk,
                                                 float* __restrict__ vv) {
    int idx = blockIdx.x * 256 + threadIdx.x;      // [b,c,n] flat, 16*512*20
    int n = idx % NNL;
    int c = (idx / NNL) % ND;
    int b = idx / (NNL * ND);
    float sk = bk[c], sv = bv[c];
    const float* lb  = l  + (size_t)b * NLC * NNL + n;
    const float* wkr = Wk + (size_t)c * NLC;
    const float* wvr = Wv + (size_t)c * NLC;
    for (int lc = 0; lc < NLC; ++lc) {
        float lv = lb[lc * NNL];
        sk = fmaf(wkr[lc], lv, sk);
        sv = fmaf(wvr[lc], lv, sv);
    }
    float m = l_mask[b * NNL + n];
    kk[idx] = sk * m * 0.044194173824159216f;      // 512^-0.5 folded
    vv[idx] = sv * m;
}

// ---------------- instance-norm stats over HW ----------------
__global__ __launch_bounds__(128) void stats_kernel(const float* __restrict__ P,
                                                    float* __restrict__ mu,
                                                    float* __restrict__ isig) {
    int b = blockIdx.x >> 2;
    int c = ((blockIdx.x & 3) * 128) + threadIdx.x;
    const float* p = P + (size_t)b * NHW * ND + c;
    float s = 0.f, s2 = 0.f;
    for (int hw = 0; hw < NHW; ++hw) {
        float v = p[(size_t)hw * ND];
        s += v; s2 = fmaf(v, v, s2);
    }
    float m   = s  * (1.0f / (float)NHW);
    float var = s2 * (1.0f / (float)NHW) - m * m;
    mu[b * ND + c]   = m;
    isig[b * ND + c] = rsqrtf(var + 1e-5f);
}

// ---------------- WMMA GEMM: out = A[57600,512] @ W[N,512]^T ----------------
// 256 threads = 8 waves; block tile M=128 x N=128; wave = 32x64 = 2x4 WMMA tiles.
// Fragments are contiguous 16B spans in LDS -> 2x ds_load_b128 each;
// 12 DS loads feed 8 WMMAs per wave per k-step.
// MODE 0: W = [Wvis;Wq] (N=1024). n<512 -> gelu -> bf16 visb. n>=512 -> f32 qp.
// MODE 1: plain, +bias -> f32 outf (lang_pre).
// MODE 2: A built on the fly = vis * instnorm(lang); epilogue gelu -> f32 d_out.
template <int MODE>
__global__ __launch_bounds__(256) void gemm_wmma(
    const u16*   __restrict__ A,
    const u16*   __restrict__ W,
    const float* __restrict__ bias,
    const float* __restrict__ bias2,
    u16*         __restrict__ outb,
    float*       __restrict__ outf,
    const float* __restrict__ langp,
    const float* __restrict__ lmu,
    const float* __restrict__ lis)
{
    __shared__ __align__(16) u16 sA[128 * LDST];
    __shared__ __align__(16) u16 sB[128 * LDST];

    const int tid  = threadIdx.x;
    const int lane = tid & 31;
    const int wave = tid >> 5;
    const int wm   = (wave & 3) * 32;     // wave M offset (4 waves over M)
    const int wn   = (wave >> 2) * 64;    // wave N offset (2 waves over N)
    const int m16  = lane & 15;
    const int hf   = (lane >> 4) & 1;
    const int rowBase = blockIdx.y * 128;
    const int colBase = blockIdx.x * 128;

    v8f acc[2][4];
    #pragma unroll
    for (int mi = 0; mi < 2; ++mi)
        #pragma unroll
        for (int ni = 0; ni < 4; ++ni)
            acc[mi][ni] = (v8f){0,0,0,0,0,0,0,0};

    const int seg = tid & 3;
    for (int k0 = 0; k0 < 512; k0 += 32) {
        const int kk = k0 + seg * 8;
        // ---- stage A tile (128x32): 2 rows of 8 bf16 per thread ----
        #pragma unroll
        for (int rr = 0; rr < 2; ++rr) {
            const int row  = (tid >> 2) + rr * 64;
            const int rowg = rowBase + row;
            uint4* dst = (uint4*)&sA[row * LDST + seg * 8];
            if constexpr (MODE != 2) {
                *dst = *(const uint4*)(&A[(size_t)rowg * ND + kk]);
            } else {
                const int bb = rowg / NHW;
                const size_t base = (size_t)rowg * ND + kk;
                const uint4  vb = *(const uint4*)(&A[base]);           // vis bf16 x8
                const float4 l0 = *(const float4*)(&langp[base]);
                const float4 l1 = *(const float4*)(&langp[base + 4]);
                const float* mup = lmu + bb * ND + kk;
                const float* isp = lis + bb * ND + kk;
                const float    lv[8] = {l0.x, l0.y, l0.z, l0.w, l1.x, l1.y, l1.z, l1.w};
                const unsigned uv[4] = {vb.x, vb.y, vb.z, vb.w};
                uint4 o;
                unsigned* op = (unsigned*)&o;
                #pragma unroll
                for (int j = 0; j < 4; ++j) {
                    float v0 = bflo(uv[j]) * ((lv[2*j]   - mup[2*j])   * isp[2*j]);
                    float v1 = bfhi(uv[j]) * ((lv[2*j+1] - mup[2*j+1]) * isp[2*j+1]);
                    op[j] = (unsigned)f2bf(v0) | ((unsigned)f2bf(v1) << 16);
                }
                *dst = o;
            }
        }
        // ---- stage B tile (128x32): 2 rows of 8 bf16 per thread ----
        #pragma unroll
        for (int rr = 0; rr < 2; ++rr) {
            const int row  = (tid >> 2) + rr * 64;
            const int wrow = colBase + row;
            uint4* dst = (uint4*)&sB[row * LDST + seg * 8];
            *dst = *(const uint4*)(&W[(size_t)wrow * ND + kk]);
        }
        __syncthreads();

        // ---- gather fragments: each = two contiguous b128 LDS loads ----
        V16 a[2], b[4];
        #pragma unroll
        for (int mi = 0; mi < 2; ++mi) {
            const int base = (wm + mi * 16 + m16) * LDST + hf * 8;
            a[mi].q[0] = *(const uint4*)(&sA[base]);
            a[mi].q[1] = *(const uint4*)(&sA[base + 16]);
        }
        #pragma unroll
        for (int ni = 0; ni < 4; ++ni) {
            const int base = (wn + ni * 16 + m16) * LDST + hf * 8;
            b[ni].q[0] = *(const uint4*)(&sB[base]);
            b[ni].q[1] = *(const uint4*)(&sB[base + 16]);
        }
        #pragma unroll
        for (int mi = 0; mi < 2; ++mi)
            #pragma unroll
            for (int ni = 0; ni < 4; ++ni)
                acc[mi][ni] = __builtin_amdgcn_wmma_f32_16x16x32_bf16(
                    false, a[mi].v, false, b[ni].v, (short)0, acc[mi][ni], false, false);
        __syncthreads();
    }

    // ---- epilogue ----
    bool qside = false;
    int  coff  = colBase;
    const float* bptr = bias;
    if constexpr (MODE == 0) {
        if (blockIdx.x >= 4) { qside = true; bptr = bias2; coff = colBase - 512; }
    }
    #pragma unroll
    for (int mi = 0; mi < 2; ++mi) {
        #pragma unroll
        for (int ni = 0; ni < 4; ++ni) {
            C8 t; t.v = acc[mi][ni];
            const int rowb = rowBase + wm + mi * 16 + hf * 8;
            const int col  = coff + wn + ni * 16 + m16;
            #pragma unroll
            for (int r = 0; r < 8; ++r) {
                const int row = rowb + r;
                const float val = t.f[r];
                if constexpr (MODE == 0) {
                    if (qside) outf[(size_t)row * ND + col] = val + bptr[col];
                    else       outb[(size_t)row * ND + col] = f2bf(gelu_exact(val + bptr[col]));
                } else if constexpr (MODE == 1) {
                    outf[(size_t)row * ND + col] = val + bptr[col];
                } else {
                    outf[(size_t)row * ND + col] = gelu_exact(val + bptr[col]);
                }
            }
        }
    }
}

// ---------------- attention (NL=20, per-pixel softmax) ----------------
__global__ __launch_bounds__(256) void attn_kernel(const float* __restrict__ qp,
                                                   const float* __restrict__ kk,
                                                   const float* __restrict__ vv,
                                                   const float* __restrict__ l_mask,
                                                   const float* __restrict__ muq,
                                                   const float* __restrict__ isq,
                                                   u16* __restrict__ outb) {
    __shared__ float ks[64 * NNL];
    __shared__ float vs[64 * NNL];
    __shared__ float mb[NNL];
    __shared__ float mus[64], iss[64];

    const int chunk = blockIdx.x % 15;
    const int bh    = blockIdx.x / 15;
    const int b     = bh >> 3;
    const int h     = bh & 7;
    const int tid   = threadIdx.x;

    for (int i = tid; i < 64 * NNL; i += 256) {
        int d = i / NNL, n = i % NNL;
        size_t ci = (size_t)(b * ND + h * 64 + d) * NNL + n;
        ks[i] = kk[ci];
        vs[i] = vv[ci];
    }
    if (tid < NNL) mb[tid] = 10000.0f * (l_mask[b * NNL + tid] - 1.0f);
    if (tid < 64) {
        mus[tid] = muq[b * ND + h * 64 + tid];
        iss[tid] = isq[b * ND + h * 64 + tid];
    }
    __syncthreads();

    const int hw = chunk * 256 + tid;
    if (hw >= NHW) return;
    const size_t row = (size_t)b * NHW + hw;

    float s[NNL];
    #pragma unroll
    for (int n = 0; n < NNL; ++n) s[n] = mb[n];

    const float4* q4 = (const float4*)(qp + row * ND + h * 64);
    #pragma unroll 4
    for (int d4 = 0; d4 < 16; ++d4) {
        float4 q = q4[d4];
        float qv[4] = {q.x, q.y, q.z, q.w};
        #pragma unroll
        for (int j = 0; j < 4; ++j) {
            int d = d4 * 4 + j;
            float qn = (qv[j] - mus[d]) * iss[d];
            #pragma unroll
            for (int n = 0; n < NNL; ++n) s[n] = fmaf(qn, ks[d * NNL + n], s[n]);
        }
    }
    float mx = s[0];
    #pragma unroll
    for (int n = 1; n < NNL; ++n) mx = fmaxf(mx, s[n]);
    float sum = 0.f;
    #pragma unroll
    for (int n = 0; n < NNL; ++n) { s[n] = __expf(s[n] - mx); sum += s[n]; }
    float inv = 1.0f / sum;
    #pragma unroll
    for (int n = 0; n < NNL; ++n) s[n] *= inv;

    u16* orow = outb + row * ND + h * 64;
    #pragma unroll 2
    for (int d2 = 0; d2 < 32; ++d2) {
        float o0 = 0.f, o1 = 0.f;
        #pragma unroll
        for (int n = 0; n < NNL; ++n) {
            o0 = fmaf(s[n], vs[(d2 * 2 + 0) * NNL + n], o0);
            o1 = fmaf(s[n], vs[(d2 * 2 + 1) * NNL + n], o1);
        }
        *(unsigned*)(&orow[d2 * 2]) = (unsigned)f2bf(o0) | ((unsigned)f2bf(o1) << 16);
    }
}

// ---------------- launch ----------------
extern "C" void kernel_launch(void* const* d_in, const int* in_sizes, int n_in,
                              void* d_out, int out_size, void* d_ws, size_t ws_size,
                              hipStream_t stream) {
    const float* x      = (const float*)d_in[0];
    const float* l      = (const float*)d_in[1];
    const float* l_mask = (const float*)d_in[2];
    const float* Wvis   = (const float*)d_in[3];
    const float* bvis   = (const float*)d_in[4];
    const float* Wq     = (const float*)d_in[5];
    const float* bq     = (const float*)d_in[6];
    const float* Wk     = (const float*)d_in[7];
    const float* bk     = (const float*)d_in[8];
    const float* Wval   = (const float*)d_in[9];
    const float* bval   = (const float*)d_in[10];
    const float* Ww     = (const float*)d_in[11];
    const float* bw     = (const float*)d_in[12];
    const float* Wmm    = (const float*)d_in[13];
    const float* bmm    = (const float*)d_in[14];
    float* out = (float*)d_out;
    char*  ws  = (char*)d_ws;

    const size_t SZ_BF = (size_t)NR * ND * 2;   // 58,982,400
    const size_t SZ_F  = (size_t)NR * ND * 4;   // 117,964,800
    size_t off = 0;
    u16*   xb    = (u16*)(ws + off);  off += SZ_BF;        // x bf16
    u16*   visb  = (u16*)(ws + off);  off += SZ_BF;        // gelu(x@Wvis^T+b) bf16
    u16*   aob   = (u16*)(ws + off);  off += SZ_BF;        // attention output bf16
    float* P     = (float*)(ws + off); off += SZ_F;        // q_pre, then lang_pre
    u16*   wcat  = (u16*)(ws + off);  off += (size_t)1024 * 512 * 2; // [Wvis;Wq] bf16
    u16*   wwb   = (u16*)(ws + off);  off += (size_t)512 * 512 * 2;
    u16*   wmb   = (u16*)(ws + off);  off += (size_t)512 * 512 * 2;
    float* kkp   = (float*)(ws + off); off += (size_t)NB * ND * NNL * 4;
    float* vvp   = (float*)(ws + off); off += (size_t)NB * ND * NNL * 4;
    float* muq   = (float*)(ws + off); off += (size_t)NB * ND * 4;
    float* isq   = (float*)(ws + off); off += (size_t)NB * ND * 4;
    float* mul_  = (float*)(ws + off); off += (size_t)NB * ND * 4;
    float* isl_  = (float*)(ws + off); off += (size_t)NB * ND * 4;

    // 1) converts
    cvt_kernel<<<NR * ND / 1024, 256, 0, stream>>>(x, xb, NR * ND);
    cvt_kernel<<<256, 256, 0, stream>>>(Wvis, wcat,                262144);
    cvt_kernel<<<256, 256, 0, stream>>>(Wq,   wcat + 262144,       262144);
    cvt_kernel<<<256, 256, 0, stream>>>(Ww,   wwb,                 262144);
    cvt_kernel<<<256, 256, 0, stream>>>(Wmm,  wmb,                 262144);
    // 2) k / v projections (+mask, +scale)
    kv_kernel<<<NB * ND * NNL / 256, 256, 0, stream>>>(l, l_mask, Wk, bk, Wval, bval, kkp, vvp);
    // 3) fused vis + q_pre GEMM (N=1024)
    gemm_wmma<0><<<dim3(8, 450), 256, 0, stream>>>(xb, wcat, bvis, bq, visb, P,
                                                   nullptr, nullptr, nullptr);
    // 4) instance-norm stats of q_pre
    stats_kernel<<<NB * 4, 128, 0, stream>>>(P, muq, isq);
    // 5) attention -> bf16
    attn_kernel<<<NB * 8 * 15, 256, 0, stream>>>(P, kkp, vvp, l_mask, muq, isq, aob);
    // 6) lang_pre = attn_out @ Ww^T + bw (reuses P)
    gemm_wmma<1><<<dim3(4, 450), 256, 0, stream>>>(aob, wwb, bw, nullptr, nullptr, P,
                                                   nullptr, nullptr, nullptr);
    // 7) instance-norm stats of lang_pre
    stats_kernel<<<NB * 4, 128, 0, stream>>>(P, mul_, isl_);
    // 8) final: gelu((vis * instnorm(lang)) @ Wmm^T + bmm) -> d_out f32
    gemm_wmma<2><<<dim3(4, 450), 256, 0, stream>>>(visb, wmb, bmm, nullptr, nullptr, out,
                                                   P, mul_, isl_);
    (void)in_sizes; (void)n_in; (void)out_size; (void)ws_size;
}
